// StyledLayer_46102178955368
// MI455X (gfx1250) — matrix operand order repaired and checked by reading, not compiled, not run on hardware
//
#include <hip/hip_runtime.h>
#include <hip/hip_bf16.h>

// ---------------- problem constants ----------------
#define BATCH 16
#define CIN   512
#define COUT  512
#define SDIM  512
#define HIN   32
#define WIN   32
#define HOUT  64
#define WOUT  64
#define KSZ   3
#define PADW  66            // HOUT+2 zero-halo frame for the staged input

#define LIN_SCALE   (0.044194173824159216f)   // 1/sqrt(512)
#define CONV_SCALE  (0.014731391274719739f)   // 1/sqrt(512*9)
#define ACT_GAIN    (1.4142135623730951f)
#define NEG_SLOPE   (0.2f)
#define EPS_DEMOD   (1e-8f)

// ---------------- WMMA fragment types ----------------
typedef __attribute__((ext_vector_type(16))) __bf16        v16bf;
typedef __attribute__((ext_vector_type(8)))  float         v8f;
typedef __attribute__((ext_vector_type(4)))  unsigned int  v4u;

union BF16x16 { v4u q[2]; v16bf v; };

// A-fragment (16x32 bf16, MxK): per lane, 16 consecutive K values starting at
// base (base = lane<16 ? 0 : 16). Two contiguous 16B loads.
__device__ __forceinline__ v16bf load_afrag(const unsigned short* p) {
    BF16x16 r;
    r.q[0] = *(const v4u*)(p);
    r.q[1] = *(const v4u*)(p + 8);
    return r.v;
}
// B-fragment (32x16 bf16, KxN): per lane, 16 consecutive K values starting at
// base (base = lane<16 ? 0 : 16). Two contiguous 16B loads.
__device__ __forceinline__ v16bf load_bfrag(const unsigned short* p) {
    BF16x16 r;
    r.q[0] = *(const v4u*)(p);
    r.q[1] = *(const v4u*)(p + 8);
    return r.v;
}

// float -> bf16 bits, round-to-nearest-even
__device__ __forceinline__ unsigned short f2bf(float f) {
    unsigned int u = __float_as_uint(f);
    u += 0x7FFFu + ((u >> 16) & 1u);
    return (unsigned short)(u >> 16);
}

// ---------------- kernel 0: zero the padded staging buffer ----------------
__global__ void zero_kernel(unsigned short* __restrict__ xs) {
    size_t id = (size_t)blockIdx.x * 256 + threadIdx.x;   // one uint4 (8 bf16) each
    v4u* p = (v4u*)xs;
    p[id] = (v4u)0u;
}

// ---------------- kernel 1: style s = w @ (lin_w*scale)^T + lin_b ----------------
__global__ void style_kernel(const float* __restrict__ w,
                             const float* __restrict__ lin_w,
                             const float* __restrict__ lin_b,
                             float* __restrict__ s) {
    int id = blockIdx.x * 256 + threadIdx.x;      // 0 .. 16*512-1
    int b = id >> 9, c = id & 511;
    const float* wr = w + (size_t)b * SDIM;
    const float* lr = lin_w + (size_t)c * SDIM;
    float acc = 0.f;
#pragma unroll 4
    for (int k = 0; k < SDIM; k += 4) {
        float4 a = *(const float4*)(wr + k);
        float4 l = *(const float4*)(lr + k);
        acc += a.x * l.x + a.y * l.y + a.z * l.z + a.w * l.w;
    }
    s[id] = acc * LIN_SCALE + lin_b[c];
}

// ---------------- kernel 2: demod[b][o] = rsqrt(sum_i s^2 * sum_kl wk^2 + eps) ----
__global__ void demod_kernel(const float* __restrict__ conv_w,
                             const float* __restrict__ s,
                             float* __restrict__ demod) {
    int o = blockIdx.x;
    int t = threadIdx.x;                           // 256 threads
    __shared__ float w2[CIN];
    __shared__ float red[256];
    for (int i = t; i < CIN; i += 256) {
        const float* p = conv_w + ((size_t)o * CIN + i) * (KSZ * KSZ);
        float sum = 0.f;
#pragma unroll
        for (int k = 0; k < KSZ * KSZ; ++k) { float v = p[k]; sum += v * v; }
        w2[i] = sum * (CONV_SCALE * CONV_SCALE);
    }
    __syncthreads();
    for (int b = 0; b < BATCH; ++b) {
        float acc = 0.f;
        for (int i = t; i < CIN; i += 256) {
            float sv = s[b * CIN + i];
            acc += sv * sv * w2[i];
        }
        red[t] = acc;
        __syncthreads();
        for (int off = 128; off > 0; off >>= 1) {
            if (t < off) red[t] += red[t + off];
            __syncthreads();
        }
        if (t == 0) demod[b * COUT + o] = rsqrtf(red[0] + EPS_DEMOD);
        __syncthreads();
    }
}

// ---------------- kernel 3: weights -> bf16, layout [tap][cout][cin] --------------
__global__ void wcvt_kernel(const float* __restrict__ conv_w,
                            unsigned short* __restrict__ wt) {
    size_t id = (size_t)blockIdx.x * 256 + threadIdx.x;   // over 9*512*512
    if (id >= (size_t)KSZ * KSZ * COUT * CIN) return;
    int i = (int)(id & 511);
    int o = (int)((id >> 9) & 511);
    int tp = (int)(id >> 18);
    float v = conv_w[((size_t)o * CIN + i) * (KSZ * KSZ) + tp] * CONV_SCALE;
    wt[id] = f2bf(v);
}

// ---------------- kernel 4: 2x bilinear upsample + modulate -> padded NHWC bf16 ---
// grid: x = spatial tile (4x4 of 16x16 output pixels), y = cin chunk (8x64), z = b
__global__ void __launch_bounds__(256)
stage_kernel(const float* __restrict__ x,
             const float* __restrict__ s,
             unsigned short* __restrict__ xs) {   // [B][66][66][512], interior only
    int b  = blockIdx.z;
    int c0 = blockIdx.y * 64;
    int ty = (blockIdx.x >> 2) * 16;   // output Y0
    int tx = (blockIdx.x & 3) * 16;    // output X0
    int iy0 = ty >> 1, ix0 = tx >> 1;  // input tile origin

    __shared__ float tile[64 * 100];   // 64 ch x 10 x 10 input patch (edge-clamped)
    int t = threadIdx.x;
    for (int idx = t; idx < 6400; idx += 256) {
        int c = idx / 100, rem = idx % 100;
        int r = rem / 10, q = rem % 10;
        int ii = min(max(iy0 - 1 + r, 0), HIN - 1);
        int jj = min(max(ix0 - 1 + q, 0), WIN - 1);
        tile[idx] = x[(((size_t)b * CIN + c0 + c) * HIN + ii) * WIN + jj];
    }
    __syncthreads();

    int c  = t & 63;
    int pg = t >> 6;                   // 4 pixel groups of 64
    float sv = s[b * CIN + c0 + c];
    const float* base = tile + c * 100;
#pragma unroll 4
    for (int p = 0; p < 64; ++p) {
        int pix = pg * 64 + p;
        int yl = pix >> 4, xl = pix & 15;
        // half-pixel 2x bilinear: even -> (0.25, 0.75), odd -> (0.75, 0.25)
        int   r0  = (yl >> 1) + (yl & 1);
        float wy0 = (yl & 1) ? 0.75f : 0.25f, wy1 = 1.0f - wy0;
        int   q0  = (xl >> 1) + (xl & 1);
        float wx0 = (xl & 1) ? 0.75f : 0.25f, wx1 = 1.0f - wx0;
        float v = wy0 * (wx0 * base[r0 * 10 + q0]       + wx1 * base[r0 * 10 + q0 + 1])
                + wy1 * (wx0 * base[(r0 + 1) * 10 + q0] + wx1 * base[(r0 + 1) * 10 + q0 + 1]);
        v *= sv;
        // +1 halo offset in both spatial dims
        xs[((((size_t)b * PADW + ty + yl + 1) * PADW) + tx + xl + 1) * CIN + c0 + c] = f2bf(v);
    }
}

// ---------------- kernel 5: implicit-GEMM 3x3 conv via WMMA bf16 ------------------
// grid: x = output row y (64), y = cout tile (8x64), z = batch (16); 8 waves/WG.
// wave(wm,wn): couts [co+wm*16, +16), pixels [wn*32, +16) and [wn*32+16, +16).
// Staged input has a 1-pixel zero halo -> all loads unconditional.
__global__ void __launch_bounds__(256)
conv_kernel(const unsigned short* __restrict__ xs,   // [B][66][66][512] bf16 (modulated, halo)
            const unsigned short* __restrict__ wt,   // [9][512][512]   bf16 (scaled)
            const float* __restrict__ demod,         // [B][512]
            const float* __restrict__ noise,         // [B][64][64]
            const float* __restrict__ nwp,           // scalar
            float* __restrict__ out) {               // [B][512][64][64]
    int y  = blockIdx.x;
    int b  = blockIdx.z;
    int co_tile = blockIdx.y * 64;

    int tid  = threadIdx.x;
    int lane = tid & 31;
    int wave = tid >> 5;
    int wm = wave & 3, wn = wave >> 2;
    int mrow = lane & 15;
    bool hiL = lane >= 16;
    int co  = co_tile + wm * 16;
    int aK  = hiL ? 16 : 0;    // A-fragment K base per lane
    int bK  = hiL ? 16 : 0;    // B-fragment K base per lane

    v8f acc0 = (v8f)0.0f;
    v8f acc1 = (v8f)0.0f;

    const unsigned short* xb = xs + (size_t)b * PADW * PADW * CIN + bK;

#pragma unroll
    for (int ky = 0; ky < 3; ++ky) {
#pragma unroll
        for (int kx = 0; kx < 3; ++kx) {
            int tap = ky * 3 + kx;
            const unsigned short* arow =
                wt + ((size_t)(tap * COUT) + co + mrow) * CIN + aK;
            // padded coords: row y+ky, col (wn*32 + mrow + kx) cover taps -1..+1
            const unsigned short* b0p =
                xb + (((size_t)(y + ky) * PADW) + wn * 32 + mrow + kx) * CIN;
            const unsigned short* b1p = b0p + (size_t)16 * CIN;
#pragma unroll 4
            for (int c0 = 0; c0 < CIN; c0 += 32) {
                v16bf av  = load_afrag(arow + c0);
                v16bf bv0 = load_bfrag(b0p + c0);
                v16bf bv1 = load_bfrag(b1p + c0);
                acc0 = __builtin_amdgcn_wmma_f32_16x16x32_bf16(
                    false, av, false, bv0, (short)0, acc0, false, false);
                acc1 = __builtin_amdgcn_wmma_f32_16x16x32_bf16(
                    false, av, false, bv1, (short)0, acc1, false, false);
            }
        }
    }

    // epilogue: demod + noise + leaky-relu * sqrt(2)
    float nw = nwp[0];
    const float* dm = demod + (size_t)b * COUT;
    int coutL = co + (hiL ? 8 : 0);   // C/D layout: VGPR r -> M = r (+8 for hi lanes)

    {
        int xg = wn * 32 + mrow;
        float nz = nw * noise[((size_t)b * HOUT + y) * WOUT + xg];
#pragma unroll
        for (int r = 0; r < 8; ++r) {
            int cg = coutL + r;
            float v = acc0[r] * dm[cg] + nz;
            v = (v >= 0.f ? v : NEG_SLOPE * v) * ACT_GAIN;
            out[(((size_t)b * COUT + cg) * HOUT + y) * WOUT + xg] = v;
        }
    }
    {
        int xg = wn * 32 + 16 + mrow;
        float nz = nw * noise[((size_t)b * HOUT + y) * WOUT + xg];
#pragma unroll
        for (int r = 0; r < 8; ++r) {
            int cg = coutL + r;
            float v = acc1[r] * dm[cg] + nz;
            v = (v >= 0.f ? v : NEG_SLOPE * v) * ACT_GAIN;
            out[(((size_t)b * COUT + cg) * HOUT + y) * WOUT + xg] = v;
        }
    }
}

// ---------------- host launcher ----------------
extern "C" void kernel_launch(void* const* d_in, const int* in_sizes, int n_in,
                              void* d_out, int out_size, void* d_ws, size_t ws_size,
                              hipStream_t stream) {
    (void)in_sizes; (void)n_in; (void)out_size; (void)ws_size;

    const float* x      = (const float*)d_in[0];  // [16,512,32,32]
    const float* w      = (const float*)d_in[1];  // [16,512]
    const float* noise  = (const float*)d_in[2];  // [16,1,64,64]
    const float* lin_w  = (const float*)d_in[3];  // [512,512]
    const float* lin_b  = (const float*)d_in[4];  // [512]
    const float* conv_w = (const float*)d_in[5];  // [512,512,3,3]
    const float* nwp    = (const float*)d_in[6];  // scalar
    float* out = (float*)d_out;                   // [16,512,64,64]

    // workspace layout (all chunks 16B-aligned)
    char* ws = (char*)d_ws;
    float* s_buf  = (float*)ws;                                        // 16*512 f32
    float* dm_buf = s_buf + BATCH * CIN;                               // 16*512 f32
    unsigned short* wt_buf = (unsigned short*)(dm_buf + BATCH * COUT); // 9*512*512 bf16
    unsigned short* xs_buf = wt_buf + (size_t)KSZ * KSZ * COUT * CIN;  // 16*66*66*512 bf16

    // 0) zero padded staging buffer (provides the conv's zero halo)
    {
        size_t nvec = (size_t)BATCH * PADW * PADW * CIN / 8;   // uint4 granules
        zero_kernel<<<(unsigned)(nvec / 256), 256, 0, stream>>>(xs_buf);
    }
    // 1) style
    style_kernel<<<(BATCH * CIN) / 256, 256, 0, stream>>>(w, lin_w, lin_b, s_buf);
    // 2) demod (needs s)
    demod_kernel<<<COUT, 256, 0, stream>>>(conv_w, s_buf, dm_buf);
    // 3) weight convert/transpose to bf16
    {
        size_t n = (size_t)KSZ * KSZ * COUT * CIN;
        wcvt_kernel<<<(unsigned)((n + 255) / 256), 256, 0, stream>>>(conv_w, wt_buf);
    }
    // 4) upsample + modulate -> padded NHWC bf16 (needs s)
    {
        dim3 grid(16, CIN / 64, BATCH);
        stage_kernel<<<grid, 256, 0, stream>>>(x, s_buf, xs_buf);
    }
    // 5) WMMA conv + fused epilogue
    {
        dim3 grid(HOUT, COUT / 64, BATCH);
        conv_kernel<<<grid, 256, 0, stream>>>(xs_buf, wt_buf, dm_buf, noise, nwp, out);
    }
}
